// SequenceDecoder_29643864277286
// MI455X (gfx1250) — matrix operand order repaired and checked by reading, not compiled, not run on hardware
//
#include <hip/hip_runtime.h>
#include <hip/hip_bf16.h>

// ---------------------------------------------------------------------------
// SequenceDecoder: restrict-last gather + Linear  ==  bf16 WMMA GEMM
//   M = B*L_OUT = 16384, N = DO = 1024, K = D = 1024
// 34.4 GFLOP vs ~132MB traffic -> run on v_wmma_f32_16x16x32_bf16.
// f32->bf16 staging: v_perm_b32 packs two pre-rounded f32 high halves in one
// VALU op (the cvt_pk_bf16_f32 builtin is not declared on this toolchain).
// ---------------------------------------------------------------------------

typedef __attribute__((ext_vector_type(16))) __bf16 v16bf;
typedef __attribute__((ext_vector_type(8)))  __bf16 v8bf;
typedef __attribute__((ext_vector_type(2)))  __bf16 v2bf;
typedef __attribute__((ext_vector_type(8)))  float  v8f;

#define B_     16
#define L_     4096
#define D_     1024
#define DO_    1024
#define LOUT_  1024
#define TILE_M 128
#define TILE_N 128
#define TILE_K 32
#define PITCH  40   // padded LDS row pitch (bf16 elems): 20-dword stride, conflict-free

union FragU { v16bf v; struct { v8bf lo, hi; } p; };
union PkU   { v2bf v; unsigned u; };

__device__ __forceinline__ unsigned bfpack2(float a, float b) {
#if __has_builtin(__builtin_amdgcn_cvt_pk_bf16_f32)
  PkU p;
  p.v = __builtin_amdgcn_cvt_pk_bf16_f32(a, b);   // dst[15:0]=cvt(a), dst[31:16]=cvt(b)
  return p.u;
#else
  // round-half-up f32 -> bf16, packed via one v_perm_b32:
  // result = { hi16(b + 0x8000), hi16(a + 0x8000) }
  unsigned ua = __float_as_uint(a) + 0x8000u;
  unsigned ub = __float_as_uint(b) + 0x8000u;
  // perm(src0=ub, src1=ua, sel): byte0=ua.b2, byte1=ua.b3, byte2=ub.b2, byte3=ub.b3
  return __builtin_amdgcn_perm(ub, ua, 0x07060302u);
#endif
}

// A fragment (16x32 bf16): lane m=lane&15 ; lanes 0-15 hold K 0-7 / 16-23,
// lanes 16-31 hold K 8-15 / 24-31  -> two ds_load_b128 from row-major tile.
__device__ __forceinline__ v16bf lds_frag_A(const __bf16* tile, int lane, int mOff) {
  const int m  = (lane & 15) + mOff;
  const int ks = (lane >> 4) * 8;
  const __bf16* p = tile + m * PITCH + ks;
  FragU f;
  f.p.lo = *(const v8bf*)(p);        // K ks..ks+7
  f.p.hi = *(const v8bf*)(p + 16);   // K 16+ks..16+ks+7
  return f.v;
}

// B fragment (32x16 bf16): lane n=lane&15 ; lanes 0-15 hold K 0-7 / 16-23,
// lanes 16-31 hold K 8-15 / 24-31. W is row-major [n][k] so columns are contiguous.
__device__ __forceinline__ v16bf lds_frag_B(const __bf16* tile, int lane, int nOff) {
  const int n  = (lane & 15) + nOff;
  const int ks = (lane >> 4) * 8;
  const __bf16* p = tile + n * PITCH + ks;
  FragU f;
  f.p.lo = *(const v8bf*)(p);
  f.p.hi = *(const v8bf*)(p + 16);
  return f.v;
}

// ---- prep: s[b] = min(sum(mask[b,:]), L_OUT) ------------------------------
__global__ void __launch_bounds__(256)
seqdec_valid_kernel(const int* __restrict__ mask, int* __restrict__ svec) {
  __shared__ int red[256];
  const int b = blockIdx.x;
  const int t = threadIdx.x;
  int sum = 0;
  for (int i = t; i < L_; i += 256) sum += mask[(size_t)b * L_ + i];
  red[t] = sum;
  __syncthreads();
  for (int off = 128; off > 0; off >>= 1) {
    if (t < off) red[t] += red[t + off];
    __syncthreads();
  }
  if (t == 0) svec[b] = red[0] < LOUT_ ? red[0] : LOUT_;
}

// ---- fused gather + GEMM + bias -------------------------------------------
__global__ void __launch_bounds__(256)
seqdec_gemm_kernel(const float* __restrict__ x, const float* __restrict__ W,
                   const float* __restrict__ bias, const int* __restrict__ svec,
                   float* __restrict__ out) {
  __shared__ __align__(16) __bf16 Ash[2][TILE_M * PITCH];
  __shared__ __align__(16) __bf16 Bsh[2][TILE_M * PITCH];

  const int tid   = threadIdx.x;
  const int lane  = tid & 31;
  const int wave  = tid >> 5;
  const int waveM = wave >> 2;   // 0..1 : 64 output rows each
  const int waveN = wave & 3;    // 0..3 : 32 output cols each

  const int nBase = blockIdx.x * TILE_N;
  const int mTile = blockIdx.y;              // 0..127
  const int bIdx  = mTile >> 3;              // batch index (8 M-tiles per batch)
  const int lBase = (mTile & 7) * TILE_M;    // position within l_output

  const int s = svec[bIdx];                  // valid length, <= 1024

  // ---- cooperative global->LDS staging: 2 threads per tile row ----
  const int row  = tid >> 1;       // 0..127
  const int half = tid & 1;
  const int col0 = half * 16;      // 16 f32 per thread per tile

  const int  l      = lBase + row;
  const bool aValid = (l < s);
  int srcL = L_ - s + l;                 // reference: clip(L - s + j, 0, L-1)
  if (srcL > L_ - 1) srcL = L_ - 1;
  const float* aSrc = x + ((size_t)bIdx * L_ + (size_t)srcL) * D_ + col0;
  const float* bSrc = W + (size_t)(nBase + row) * D_ + col0;

  float4 aReg[4], bReg[4];

  auto gload = [&](int ko) {
    const float4* ap = (const float4*)(aSrc + ko);
    const float4* bp = (const float4*)(bSrc + ko);
#pragma unroll
    for (int j = 0; j < 4; ++j) {
      float4 t = {0.f, 0.f, 0.f, 0.f};
      if (aValid) t = ap[j];           // exec-masked b128 load; zero-pad rows l>=s
      aReg[j] = t;
      bReg[j] = bp[j];
    }
  };

  auto sstore = [&](int buf) {
    __bf16* ad = &Ash[buf][row * PITCH + col0];
    __bf16* bd = &Bsh[buf][row * PITCH + col0];
    uint4 a0 = { bfpack2(aReg[0].x, aReg[0].y), bfpack2(aReg[0].z, aReg[0].w),
                 bfpack2(aReg[1].x, aReg[1].y), bfpack2(aReg[1].z, aReg[1].w) };
    uint4 a1 = { bfpack2(aReg[2].x, aReg[2].y), bfpack2(aReg[2].z, aReg[2].w),
                 bfpack2(aReg[3].x, aReg[3].y), bfpack2(aReg[3].z, aReg[3].w) };
    uint4 b0 = { bfpack2(bReg[0].x, bReg[0].y), bfpack2(bReg[0].z, bReg[0].w),
                 bfpack2(bReg[1].x, bReg[1].y), bfpack2(bReg[1].z, bReg[1].w) };
    uint4 b1 = { bfpack2(bReg[2].x, bReg[2].y), bfpack2(bReg[2].z, bReg[2].w),
                 bfpack2(bReg[3].x, bReg[3].y), bfpack2(bReg[3].z, bReg[3].w) };
    *(uint4*)(ad)     = a0;  *(uint4*)(ad + 8) = a1;   // 2x ds_store_b128
    *(uint4*)(bd)     = b0;  *(uint4*)(bd + 8) = b1;
  };

  v8f acc[4][2];
#pragma unroll
  for (int mf = 0; mf < 4; ++mf)
#pragma unroll
    for (int nf = 0; nf < 2; ++nf)
      acc[mf][nf] = (v8f){0.f, 0.f, 0.f, 0.f, 0.f, 0.f, 0.f, 0.f};

  // software pipeline: prefetch K-step k+1 from global while WMMA on k
  gload(0);
  sstore(0);

  const int KT = D_ / TILE_K;  // 32 iterations
  for (int kt = 0; kt < KT; ++kt) {
    __syncthreads();                       // buf[kt&1] stores visible to all waves
    const int buf = kt & 1;
    if (kt + 1 < KT) gload((kt + 1) * TILE_K);

    v16bf afr[4], bfr[2];
#pragma unroll
    for (int mf = 0; mf < 4; ++mf)
      afr[mf] = lds_frag_A(&Ash[buf][0], lane, waveM * 64 + mf * 16);
#pragma unroll
    for (int nf = 0; nf < 2; ++nf)
      bfr[nf] = lds_frag_B(&Bsh[buf][0], lane, waveN * 32 + nf * 16);

#pragma unroll
    for (int mf = 0; mf < 4; ++mf)
#pragma unroll
      for (int nf = 0; nf < 2; ++nf)
        acc[mf][nf] = __builtin_amdgcn_wmma_f32_16x16x32_bf16(
            false, afr[mf], false, bfr[nf],
            (short)0, acc[mf][nf], false, false);

    if (kt + 1 < KT) sstore(buf ^ 1);      // fills the other buffer; no race
  }

  // ---- epilogue: + bias, f32 store (C layout: VGPR i -> M, lane&15 -> N) ----
  const int mWave = mTile * TILE_M + waveM * 64;
  const int nWave = nBase + waveN * 32;
  const int nl    = lane & 15;
  const int mh    = (lane >> 4) * 8;
#pragma unroll
  for (int nf = 0; nf < 2; ++nf) {
    const float bv = bias[nWave + nf * 16 + nl];
#pragma unroll
    for (int mf = 0; mf < 4; ++mf) {
#pragma unroll
      for (int i = 0; i < 8; ++i) {
        const int r = mWave + mf * 16 + mh + i;
        out[(size_t)r * DO_ + (nWave + nf * 16 + nl)] = acc[mf][nf][i] + bv;
      }
    }
  }
}

extern "C" void kernel_launch(void* const* d_in, const int* in_sizes, int n_in,
                              void* d_out, int out_size, void* d_ws, size_t ws_size,
                              hipStream_t stream) {
  const float* x    = (const float*)d_in[0];   // (16, 4096, 1024) f32
  const int*   mask = (const int*)d_in[1];     // (16, 4096) i32
  const float* W    = (const float*)d_in[2];   // (1024, 1024) f32
  const float* bias = (const float*)d_in[3];   // (1024,) f32
  float*       out  = (float*)d_out;           // (16, 1024, 1024) f32
  int*         svec = (int*)d_ws;              // 16 ints of scratch

  seqdec_valid_kernel<<<B_, 256, 0, stream>>>(mask, svec);
  seqdec_gemm_kernel<<<dim3(DO_ / TILE_N, (B_ * LOUT_) / TILE_M), 256, 0, stream>>>(
      x, W, bias, svec, out);
}
